// MixtureOfExperts_43241730737080
// MI455X (gfx1250) — matrix operand order repaired and checked by reading, not compiled, not run on hardware
//
#include <hip/hip_runtime.h>
#include <hip/hip_bf16.h>

#define T_TOKENS 8192
#define MOE_DIM 1024
#define MOE_HID 2048
#define NEXP 8
#define NSLOTS (T_TOKENS * 2)

typedef __attribute__((ext_vector_type(16))) __bf16 v16bf;
typedef __attribute__((ext_vector_type(8)))  float  v8f;
typedef int v4i __attribute__((vector_size(16)));

union FragBF { uint4 q[2]; v16bf v; };

#if __has_builtin(__builtin_amdgcn_global_load_async_to_lds_b128)
#define MOE_ASYNC 1
#else
#define MOE_ASYNC 0
#endif

__device__ __forceinline__ void async_cp16(unsigned short* ldst,
                                           const unsigned short* gsrc) {
#if MOE_ASYNC
    __builtin_amdgcn_global_load_async_to_lds_b128(
        (__attribute__((address_space(1))) v4i*)gsrc,
        (__attribute__((address_space(3))) v4i*)ldst,
        0, 0);
#else
    *(uint4*)ldst = *(const uint4*)gsrc;
#endif
}

template <int N>
__device__ __forceinline__ void wait_asynccnt() {
#if MOE_ASYNC
#if __has_builtin(__builtin_amdgcn_s_wait_asynccnt)
    __builtin_amdgcn_s_wait_asynccnt((short)N);
#else
    asm volatile("s_wait_asynccnt %0" :: "i"(N) : "memory");
#endif
#endif
}

__device__ __forceinline__ unsigned short f2bf(float f) {
    unsigned int u = __float_as_uint(f);
    unsigned int r = (u + 0x7FFFu + ((u >> 16) & 1u)) >> 16;  // RNE
    return (unsigned short)r;
}

// ---------------- small kernels ----------------

__global__ void moe_zero_counts(int* counts) {
    if (threadIdx.x < NEXP) counts[threadIdx.x] = 0;
}

__global__ void moe_convert_bf16(const float* __restrict__ src,
                                 unsigned short* __restrict__ dst, int n) {
    int i = blockIdx.x * blockDim.x + threadIdx.x;
    int stride = gridDim.x * blockDim.x;
    for (; i < n; i += stride) dst[i] = f2bf(src[i]);
}

__global__ void moe_router(const float* __restrict__ x,
                           const float* __restrict__ rw,
                           int* __restrict__ eids, float* __restrict__ wts,
                           int* __restrict__ counts) {
    int t = blockIdx.x * blockDim.x + threadIdx.x;
    if (t >= T_TOKENS) return;
    const float* xr = x + (size_t)t * MOE_DIM;
    float l[NEXP];
#pragma unroll
    for (int e = 0; e < NEXP; ++e) l[e] = 0.0f;
    for (int d = 0; d < MOE_DIM; ++d) {
        float xv = xr[d];
#pragma unroll
        for (int e = 0; e < NEXP; ++e) l[e] += xv * rw[e * MOE_DIM + d];
    }
    int a = 0;
#pragma unroll
    for (int e = 1; e < NEXP; ++e) if (l[e] > l[a]) a = e;
    int b = (a == 0) ? 1 : 0;
#pragma unroll
    for (int e = 0; e < NEXP; ++e) if (e != a && l[e] > l[b]) b = e;
    // renormalized top-2 softmax weights
    float wa = 1.0f / (1.0f + __expf(l[b] - l[a]));
    float wb = 1.0f - wa;
    eids[2 * t] = a;      eids[2 * t + 1] = b;
    wts[2 * t]  = wa;     wts[2 * t + 1]  = wb;
    atomicAdd(&counts[a], 1);
    atomicAdd(&counts[b], 1);
}

__global__ void moe_offsets(const int* __restrict__ counts, int* __restrict__ offs) {
    if (threadIdx.x == 0 && blockIdx.x == 0) {
        int acc = 0;
        for (int e = 0; e < NEXP; ++e) { offs[e] = acc; acc += counts[e]; }
        offs[NEXP] = acc;   // == NSLOTS
    }
}

// Deterministic slot assignment: stable (token-id order) position within each
// expert segment, computed by a single-block LDS prefix scan.
__global__ __launch_bounds__(256) void moe_assign(const int* __restrict__ eids,
                                                  const int* __restrict__ offs,
                                                  int* __restrict__ perm,
                                                  int* __restrict__ slotOf) {
    __shared__ int baseS[NEXP];
    __shared__ int sc[256];
    int tid = threadIdx.x;
    if (tid < NEXP) baseS[tid] = offs[tid];
    __syncthreads();
    for (int chunk = 0; chunk < T_TOKENS; chunk += 256) {
        int t = chunk + tid;
        int e0 = eids[2 * t], e1 = eids[2 * t + 1];
        for (int e = 0; e < NEXP; ++e) {
            int ind = (e0 == e ? 1 : 0) + (e1 == e ? 1 : 0);
            __syncthreads();
            sc[tid] = ind;
            __syncthreads();
            for (int off = 1; off < 256; off <<= 1) {
                int cur = sc[tid];
                int add = (tid >= off) ? sc[tid - off] : 0;
                __syncthreads();
                sc[tid] = cur + add;
                __syncthreads();
            }
            int excl = sc[tid] - ind;
            int tot  = sc[255];
            int slot = baseS[e] + excl;
            if (e0 == e)      { perm[slot] = t; slotOf[2 * t]     = slot; }
            else if (e1 == e) { perm[slot] = t; slotOf[2 * t + 1] = slot; }
            __syncthreads();
            if (tid == 0) baseS[e] += tot;
        }
    }
}

__global__ void moe_gather(const float* __restrict__ x,
                           const int* __restrict__ perm,
                           unsigned short* __restrict__ xb) {
    int slot = blockIdx.x;
    int t = perm[slot];
    const float* src = x + (size_t)t * MOE_DIM;
    unsigned short* dst = xb + (size_t)slot * MOE_DIM;
    for (int d = threadIdx.x; d < MOE_DIM; d += blockDim.x) dst[d] = f2bf(src[d]);
}

// ---------------- GEMM 1&3 fused: H = silu(X W1^T) * (X W3^T) ----------------
// block: 256 threads (8 waves). Block tile 128(M) x 128(N); wave tile 64x32.
// Double-buffered async global->LDS staging (ASYNCcnt).
__global__ __launch_bounds__(256) void moe_gemm13(
    const unsigned short* __restrict__ xb,
    const unsigned short* __restrict__ w1b,
    const unsigned short* __restrict__ w3b,
    unsigned short* __restrict__ hb,
    const int* __restrict__ offs,
    const int* __restrict__ counts) {
    const int e = blockIdx.z;
    const int cnt = counts[e];
    const int mtile = blockIdx.x;               // 0..127
    if (mtile * 128 >= cnt) return;
    const int ntile = blockIdx.y;               // 0..15
    const int tid = threadIdx.x;
    const int wave = tid >> 5;
    const int lane = tid & 31;
    const int lrow = lane & 15;
    const int khalf = lane >> 4;
    const int wm = wave & 1;                    // M offset wm*64
    const int wn = wave >> 1;                   // N offset wn*32
    const int rowbase = offs[e] + mtile * 128;
    const int nbase = ntile * 128;

    __shared__ unsigned short A_s[2][128][40];
    __shared__ unsigned short B1_s[2][128][40];
    __shared__ unsigned short B3_s[2][128][40];

    const unsigned short* w1e = w1b + (size_t)e * MOE_HID * MOE_DIM;
    const unsigned short* w3e = w3b + (size_t)e * MOE_HID * MOE_DIM;

    v8f acc1[4][2], acc3[4][2];
    {
        v8f z = {};
#pragma unroll
        for (int m = 0; m < 4; ++m)
#pragma unroll
            for (int n = 0; n < 2; ++n) { acc1[m][n] = z; acc3[m][n] = z; }
    }

    // issue one K-step's tiles (A, B1, B3) into buffer `buf` (6 async/wave)
    const int r0 = tid >> 2, c0 = tid & 3;      // covers idx = tid and tid+256
#define GEMM13_ISSUE(buf, k0)                                                      \
    {                                                                              \
        int r = r0, c = c0;                                                        \
        int grow = rowbase + r; if (grow > NSLOTS - 1) grow = NSLOTS - 1;          \
        async_cp16(&A_s[buf][r][c * 8],  xb  + (size_t)grow * MOE_DIM + (k0) + c * 8); \
        async_cp16(&B1_s[buf][r][c * 8], w1e + (size_t)(nbase + r) * MOE_DIM + (k0) + c * 8); \
        async_cp16(&B3_s[buf][r][c * 8], w3e + (size_t)(nbase + r) * MOE_DIM + (k0) + c * 8); \
        r = (tid + 256) >> 2; c = (tid + 256) & 3;                                 \
        grow = rowbase + r; if (grow > NSLOTS - 1) grow = NSLOTS - 1;              \
        async_cp16(&A_s[buf][r][c * 8],  xb  + (size_t)grow * MOE_DIM + (k0) + c * 8); \
        async_cp16(&B1_s[buf][r][c * 8], w1e + (size_t)(nbase + r) * MOE_DIM + (k0) + c * 8); \
        async_cp16(&B3_s[buf][r][c * 8], w3e + (size_t)(nbase + r) * MOE_DIM + (k0) + c * 8); \
    }

    GEMM13_ISSUE(0, 0)

    for (int k0 = 0; k0 < MOE_DIM; k0 += 32) {
        const int cur = (k0 >> 5) & 1;
        if (k0 + 32 < MOE_DIM) {
            GEMM13_ISSUE(cur ^ 1, k0 + 32)
            wait_asynccnt<6>();   // in-order: the 6 loads of `cur` have landed
        } else {
            wait_asynccnt<0>();
        }
        __syncthreads();

        v16bf afr[4];
#pragma unroll
        for (int mt = 0; mt < 4; ++mt) {
            const unsigned short* ar = &A_s[cur][wm * 64 + mt * 16 + lrow][0];
            FragBF f;
            f.q[0] = *(const uint4*)(ar + khalf * 16);
            f.q[1] = *(const uint4*)(ar + khalf * 16 + 8);
            afr[mt] = f.v;
        }
#pragma unroll
        for (int nt = 0; nt < 2; ++nt) {
            const unsigned short* b1r = &B1_s[cur][wn * 32 + nt * 16 + lrow][0];
            FragBF f1;
            f1.q[0] = *(const uint4*)(b1r + khalf * 16);
            f1.q[1] = *(const uint4*)(b1r + khalf * 16 + 8);
            const unsigned short* b3r = &B3_s[cur][wn * 32 + nt * 16 + lrow][0];
            FragBF f3;
            f3.q[0] = *(const uint4*)(b3r + khalf * 16);
            f3.q[1] = *(const uint4*)(b3r + khalf * 16 + 8);
#pragma unroll
            for (int mt = 0; mt < 4; ++mt) {
                acc1[mt][nt] = __builtin_amdgcn_wmma_f32_16x16x32_bf16(
                    false, afr[mt], false, f1.v, (short)0, acc1[mt][nt], false, false);
                acc3[mt][nt] = __builtin_amdgcn_wmma_f32_16x16x32_bf16(
                    false, afr[mt], false, f3.v, (short)0, acc3[mt][nt], false, false);
            }
        }
        __syncthreads();
    }
#undef GEMM13_ISSUE

#pragma unroll
    for (int mt = 0; mt < 4; ++mt) {
#pragma unroll
        for (int nt = 0; nt < 2; ++nt) {
            v8f a = acc1[mt][nt], b = acc3[mt][nt];
#pragma unroll
            for (int v = 0; v < 8; ++v) {
                int m = wm * 64 + mt * 16 + khalf * 8 + v;   // C layout: M = v + 8*khalf
                if (mtile * 128 + m < cnt) {
                    float x1 = a[v], x3 = b[v];
                    float h = (x1 / (1.0f + __expf(-x1))) * x3;   // silu(x1)*x3
                    int gcol = nbase + wn * 32 + nt * 16 + lrow;  // C layout: N = lane%16
                    hb[(size_t)(rowbase + m) * MOE_HID + gcol] = f2bf(h);
                }
            }
        }
    }
}

// ---------------- GEMM 2: Y = H W2^T  (fp32 out) ----------------
__global__ __launch_bounds__(256) void moe_gemm2(
    const unsigned short* __restrict__ hb,
    const unsigned short* __restrict__ w2b,
    float* __restrict__ yb,
    const int* __restrict__ offs,
    const int* __restrict__ counts) {
    const int e = blockIdx.z;
    const int cnt = counts[e];
    const int mtile = blockIdx.x;               // 0..127
    if (mtile * 128 >= cnt) return;
    const int ntile = blockIdx.y;               // 0..7
    const int tid = threadIdx.x;
    const int wave = tid >> 5;
    const int lane = tid & 31;
    const int lrow = lane & 15;
    const int khalf = lane >> 4;
    const int wm = wave & 1;
    const int wn = wave >> 1;
    const int rowbase = offs[e] + mtile * 128;
    const int nbase = ntile * 128;

    __shared__ unsigned short A_s[2][128][40];
    __shared__ unsigned short B_s[2][128][40];

    const unsigned short* w2e = w2b + (size_t)e * MOE_DIM * MOE_HID;

    v8f acc[4][2];
    {
        v8f z = {};
#pragma unroll
        for (int m = 0; m < 4; ++m)
#pragma unroll
            for (int n = 0; n < 2; ++n) acc[m][n] = z;
    }

    const int r0 = tid >> 2, c0 = tid & 3;
#define GEMM2_ISSUE(buf, k0)                                                       \
    {                                                                              \
        int r = r0, c = c0;                                                        \
        int grow = rowbase + r; if (grow > NSLOTS - 1) grow = NSLOTS - 1;          \
        async_cp16(&A_s[buf][r][c * 8], hb  + (size_t)grow * MOE_HID + (k0) + c * 8); \
        async_cp16(&B_s[buf][r][c * 8], w2e + (size_t)(nbase + r) * MOE_HID + (k0) + c * 8); \
        r = (tid + 256) >> 2; c = (tid + 256) & 3;                                 \
        grow = rowbase + r; if (grow > NSLOTS - 1) grow = NSLOTS - 1;              \
        async_cp16(&A_s[buf][r][c * 8], hb  + (size_t)grow * MOE_HID + (k0) + c * 8); \
        async_cp16(&B_s[buf][r][c * 8], w2e + (size_t)(nbase + r) * MOE_HID + (k0) + c * 8); \
    }

    GEMM2_ISSUE(0, 0)

    for (int k0 = 0; k0 < MOE_HID; k0 += 32) {
        const int cur = (k0 >> 5) & 1;
        if (k0 + 32 < MOE_HID) {
            GEMM2_ISSUE(cur ^ 1, k0 + 32)
            wait_asynccnt<4>();
        } else {
            wait_asynccnt<0>();
        }
        __syncthreads();

        v16bf afr[4];
#pragma unroll
        for (int mt = 0; mt < 4; ++mt) {
            const unsigned short* ar = &A_s[cur][wm * 64 + mt * 16 + lrow][0];
            FragBF f;
            f.q[0] = *(const uint4*)(ar + khalf * 16);
            f.q[1] = *(const uint4*)(ar + khalf * 16 + 8);
            afr[mt] = f.v;
        }
#pragma unroll
        for (int nt = 0; nt < 2; ++nt) {
            const unsigned short* br = &B_s[cur][wn * 32 + nt * 16 + lrow][0];
            FragBF fb;
            fb.q[0] = *(const uint4*)(br + khalf * 16);
            fb.q[1] = *(const uint4*)(br + khalf * 16 + 8);
#pragma unroll
            for (int mt = 0; mt < 4; ++mt) {
                acc[mt][nt] = __builtin_amdgcn_wmma_f32_16x16x32_bf16(
                    false, afr[mt], false, fb.v, (short)0, acc[mt][nt], false, false);
            }
        }
        __syncthreads();
    }
#undef GEMM2_ISSUE

#pragma unroll
    for (int mt = 0; mt < 4; ++mt) {
#pragma unroll
        for (int nt = 0; nt < 2; ++nt) {
            v8f a = acc[mt][nt];
#pragma unroll
            for (int v = 0; v < 8; ++v) {
                int m = wm * 64 + mt * 16 + khalf * 8 + v;
                if (mtile * 128 + m < cnt) {
                    int gcol = nbase + wn * 32 + nt * 16 + lrow;
                    yb[(size_t)(rowbase + m) * MOE_DIM + gcol] = a[v];
                }
            }
        }
    }
}

__global__ void moe_combine(const float* __restrict__ yb,
                            const int* __restrict__ slotOf,
                            const float* __restrict__ wts,
                            float* __restrict__ out) {
    int t = blockIdx.x;
    int s0 = slotOf[2 * t], s1 = slotOf[2 * t + 1];
    float w0 = wts[2 * t], w1 = wts[2 * t + 1];
    const float* y0 = yb + (size_t)s0 * MOE_DIM;
    const float* y1 = yb + (size_t)s1 * MOE_DIM;
    float* o = out + (size_t)t * MOE_DIM;
    for (int d = threadIdx.x; d < MOE_DIM; d += blockDim.x)
        o[d] = w0 * y0[d] + w1 * y1[d];
}

// ---------------- launch ----------------

extern "C" void kernel_launch(void* const* d_in, const int* in_sizes, int n_in,
                              void* d_out, int out_size, void* d_ws, size_t ws_size,
                              hipStream_t stream) {
    const float* x  = (const float*)d_in[0];   // [T, D]
    const float* rw = (const float*)d_in[1];   // [E, D]
    const float* w1 = (const float*)d_in[2];   // [E, H, D]
    const float* w2 = (const float*)d_in[3];   // [E, D, H]
    const float* w3 = (const float*)d_in[4];   // [E, H, D]
    float* out = (float*)d_out;                // [T, D]

    char* ws = (char*)d_ws;
    size_t o = 0;
    unsigned short* w1b = (unsigned short*)(ws + o); o += (size_t)NEXP * MOE_HID * MOE_DIM * 2;
    unsigned short* w3b = (unsigned short*)(ws + o); o += (size_t)NEXP * MOE_HID * MOE_DIM * 2;
    unsigned short* w2b = (unsigned short*)(ws + o); o += (size_t)NEXP * MOE_DIM * MOE_HID * 2;
    unsigned short* xb  = (unsigned short*)(ws + o); o += (size_t)NSLOTS * MOE_DIM * 2;
    unsigned short* hb  = (unsigned short*)(ws + o); o += (size_t)NSLOTS * MOE_HID * 2;
    float* yb           = (float*)(ws + o);          o += (size_t)NSLOTS * MOE_DIM * 4;
    int* counts         = (int*)(ws + o);            o += 128;
    int* offs           = (int*)(ws + o);            o += 128;
    int* eids           = (int*)(ws + o);            o += (size_t)T_TOKENS * 2 * 4;
    float* wts          = (float*)(ws + o);          o += (size_t)T_TOKENS * 2 * 4;
    int* slotOf         = (int*)(ws + o);            o += (size_t)T_TOKENS * 2 * 4;
    int* perm           = (int*)(ws + o);            o += (size_t)NSLOTS * 4;

    const int nW = NEXP * MOE_HID * MOE_DIM;   // 16,777,216 per matrix

    hipLaunchKernelGGL(moe_zero_counts, dim3(1), dim3(32), 0, stream, counts);
    hipLaunchKernelGGL(moe_convert_bf16, dim3(8192), dim3(256), 0, stream, w1, w1b, nW);
    hipLaunchKernelGGL(moe_convert_bf16, dim3(8192), dim3(256), 0, stream, w3, w3b, nW);
    hipLaunchKernelGGL(moe_convert_bf16, dim3(8192), dim3(256), 0, stream, w2, w2b, nW);
    hipLaunchKernelGGL(moe_router, dim3(T_TOKENS / 256), dim3(256), 0, stream,
                       x, rw, eids, wts, counts);
    hipLaunchKernelGGL(moe_offsets, dim3(1), dim3(32), 0, stream, counts, offs);
    hipLaunchKernelGGL(moe_assign, dim3(1), dim3(256), 0, stream, eids, offs, perm, slotOf);
    hipLaunchKernelGGL(moe_gather, dim3(NSLOTS), dim3(256), 0, stream, x, perm, xb);
    hipLaunchKernelGGL(moe_gemm13, dim3(128, MOE_HID / 128, NEXP), dim3(256), 0, stream,
                       xb, w1b, w3b, hb, offs, counts);
    hipLaunchKernelGGL(moe_gemm2, dim3(128, MOE_DIM / 128, NEXP), dim3(256), 0, stream,
                       hb, w2b, yb, offs, counts);
    hipLaunchKernelGGL(moe_combine, dim3(T_TOKENS), dim3(256), 0, stream,
                       yb, slotOf, wts, out);
}
